// Grader_86552180949555
// MI455X (gfx1250) — matrix operand order repaired and checked by reading, not compiled, not run on hardware
//
#include <hip/hip_runtime.h>
#include <hip/hip_bf16.h>
#include <math.h>

// Problem constants (from reference)
constexpr int B = 32, S = 2048, E = 300, H = 256, G = 5;
constexpr int KP  = 320;  // E padded to multiple of 32 (WMMA f16 K)
constexpr int LDA = 328;  // LDS stride (halves) for staged emb tile (bank-conflict pad)
constexpr int LDH = 264;  // LDS stride (halves) for h state (bank-conflict pad)

typedef __attribute__((ext_vector_type(16))) _Float16 v16h;
typedef __attribute__((ext_vector_type(8)))  _Float16 v8h;
typedef __attribute__((ext_vector_type(8)))  float    v8f;

// Native CDNA5 TANH trans-op when the builtin exists; OCML fallback otherwise.
__device__ __forceinline__ float fast_tanh(float x) {
#if __has_builtin(__builtin_amdgcn_tanhf)
  return __builtin_amdgcn_tanhf(x);
#else
  return tanhf(x);
#endif
}

// ---------------- workspace layout (bytes) ----------------
constexpr size_t OFF_WIH  = 0;                                   // [2][H][KP] f16
constexpr size_t OFF_WHH  = OFF_WIH + (size_t)2 * H * KP * 2;    // [2][H][H]  f16
constexpr size_t OFF_BIAS = OFF_WHH + (size_t)2 * H * H * 2;     // [2][H]     f32
constexpr size_t OFF_POOL = OFF_BIAS + (size_t)2 * H * 4;        // [2][H][B]  f32
constexpr size_t OFF_XP   = 657408;                              // [2][S][H][B] f16

// ============ Kernel 1: convert / pack parameters to f16 ============
__global__ void grd_pack_params(const float* __restrict__ WihF, const float* __restrict__ WhhF,
                                const float* __restrict__ bihF, const float* __restrict__ bhhF,
                                const float* __restrict__ WihB, const float* __restrict__ WhhB,
                                const float* __restrict__ bihB, const float* __restrict__ bhhB,
                                _Float16* __restrict__ wih16, _Float16* __restrict__ whh16,
                                float* __restrict__ bias) {
  int i = blockIdx.x * blockDim.x + threadIdx.x;
  if (i < 2 * H * KP) {                    // W_ih, K-padded with zeros
    int d = i / (H * KP);
    int rem = i % (H * KP);
    int n = rem / KP, k = rem % KP;
    const float* W = d ? WihB : WihF;
    wih16[i] = (k < E) ? (_Float16)W[n * E + k] : (_Float16)0.f;
  }
  if (i < 2 * H * H) {                     // W_hh
    int d = i / (H * H);
    int rem = i % (H * H);
    const float* W = d ? WhhB : WhhF;
    whh16[i] = (_Float16)W[rem];
  }
  if (i < 2 * H) {                         // folded bias b_ih + b_hh
    int d = i / H, n = i % H;
    bias[i] = d ? (bihB[n] + bhhB[n]) : (bihF[n] + bhhF[n]);
  }
}

// ============ Kernel 2: embedding gather + input projection (WMMA GEMM) ============
// Block = one 16-row M-tile of x_tm (fixed s, 16 consecutive b). 8 waves:
// wave w -> direction d = w/4, N-tiles 4*(w%4)..+3. K = 10 tiles of 32.
__global__ __launch_bounds__(256) void grd_embed_proj(
    const int* __restrict__ tok, const float* __restrict__ emb,
    const _Float16* __restrict__ wih16, const float* __restrict__ bias,
    _Float16* __restrict__ xproj) {
  __shared__ __align__(32) _Float16 tile[16 * LDA];
  __shared__ int toks[16];
  const int t = threadIdx.x;
  const int mt = blockIdx.x;             // 0..4095
  const int s = mt >> 1;
  const int bbase = (mt & 1) * 16;

  if (t < 16) toks[t] = tok[(bbase + t) * S + s];
  __syncthreads();
  for (int i = t; i < 16 * KP; i += 256) {      // stage gathered rows as f16
    int row = i / KP, k = i % KP;
    tile[row * LDA + k] =
        (k < E) ? (_Float16)emb[(size_t)toks[row] * E + k] : (_Float16)0.f;
  }
  __syncthreads();

  const int w = t >> 5, lane = t & 31;
  const int g = lane >> 4, ln = lane & 15;
  const int d = w >> 2;
  const int ntbase = (w & 3) * 4;

  v8f acc[4] = {{}, {}, {}, {}};
  const _Float16* wbase = wih16 + (size_t)d * H * KP;
#pragma unroll
  for (int kt = 0; kt < KP / 32; ++kt) {
    // A fragment: lane ln, K halves at kt*32 + 16g (v0..3) and +8 (v4..7)
    const _Float16* ap = &tile[ln * LDA + kt * 32 + 16 * g];
    union { v16h v; v8h h[2]; } a;
    a.h[0] = *(const v8h*)ap;
    a.h[1] = *(const v8h*)(ap + 8);
    v16h bf[4];
#pragma unroll
    for (int q = 0; q < 4; ++q) {          // issue all 4 B loads first
      const int n = (ntbase + q) * 16 + ln;
      bf[q] = *(const v16h*)&wbase[(size_t)n * KP + kt * 32 + 16 * g];
    }
#pragma unroll
    for (int q = 0; q < 4; ++q)
      acc[q] = __builtin_amdgcn_wmma_f32_16x16x32_f16(
          false, a.v, false, bf[q], (short)0, acc[q], false, false);
  }
  // Store with folded bias: lane holds N=n fixed, M rows b0..b0+7 contiguous.
#pragma unroll
  for (int q = 0; q < 4; ++q) {
    const int n = (ntbase + q) * 16 + ln;
    const float bv = bias[d * H + n];
    v8h o;
#pragma unroll
    for (int r = 0; r < 8; ++r) o[r] = (_Float16)(acc[q][r] + bv);
    const size_t off = (((size_t)d * S + s) * H + n) * B + (bbase + 8 * g);
    *(v8h*)&xproj[off] = o;
  }
}

// ============ Kernel 3: persistent recurrent scan (2 blocks = 2 directions) ============
// 512 threads = 16 waves; wave w owns N-tile w. W_hh B-fragments (8 x v16h)
// stay resident in VGPRs across all 2048 steps. h double-buffered in LDS.
__global__ __launch_bounds__(512) void grd_rnn_scan(
    const _Float16* __restrict__ whh16, const _Float16* __restrict__ xproj,
    float* __restrict__ pool) {
  __shared__ __align__(32) _Float16 hbuf[2][32 * LDH];
  const int d = blockIdx.x;
  const int t = threadIdx.x;
  const int w = t >> 5, lane = t & 31;
  const int g = lane >> 4, ln = lane & 15;
  const int n = w * 16 + ln;  // hidden column owned by this lane

  // Resident B fragments of W_hh^T (B[k][n] = W_hh[n][k]; row n, 16 K halves @16g)
  v16h bfrag[8];
  const _Float16* wb = whh16 + (size_t)d * H * H + (size_t)n * H;
#pragma unroll
  for (int kt = 0; kt < 8; ++kt)
    bfrag[kt] = *(const v16h*)&wb[kt * 32 + 16 * g];

  for (int i = t; i < 32 * LDH; i += 512) hbuf[0][i] = (_Float16)0.f;  // h0 = 0
  v8f pool0 = {}, pool1 = {};
  __syncthreads();

  const _Float16* xbase = xproj + (size_t)d * S * H * B;
  for (int step = 0; step < S; ++step) {
    const int s = d ? (S - 1 - step) : step;           // backward dir runs reversed
    const _Float16* cur = hbuf[step & 1];
    _Float16* nxt = hbuf[(step & 1) ^ 1];

    // Issue this step's xp global loads + next step's prefetch FIRST so their
    // latency hides under the WMMA chain below.
    const _Float16* xp = &xbase[((size_t)s * H + n) * B];
    const v8h xp0 = *(const v8h*)&xp[8 * g];
    const v8h xp1 = *(const v8h*)&xp[16 + 8 * g];
    if (step + 1 < S) {
      const int s2 = d ? (S - 2 - step) : step + 1;
      __builtin_prefetch((const void*)&xbase[((size_t)s2 * H + n) * B], 0, 0);
    }

    v8f acc0 = {}, acc1 = {};
#pragma unroll
    for (int kc = 0; kc < 2; ++kc) {       // chunks of 4 K-tiles: cluster the
      union { v16h v; v8h h[2]; } a0[4], a1[4];        // ds loads ahead of WMMAs
#pragma unroll
      for (int j = 0; j < 4; ++j) {
        const int kt = kc * 4 + j;
        const _Float16* ap0 = &cur[ln * LDH + kt * 32 + 16 * g];         // b = ln
        const _Float16* ap1 = &cur[(16 + ln) * LDH + kt * 32 + 16 * g];  // b = 16+ln
        a0[j].h[0] = *(const v8h*)ap0;  a0[j].h[1] = *(const v8h*)(ap0 + 8);
        a1[j].h[0] = *(const v8h*)ap1;  a1[j].h[1] = *(const v8h*)(ap1 + 8);
      }
#pragma unroll
      for (int j = 0; j < 4; ++j) {
        const int kt = kc * 4 + j;
        acc0 = __builtin_amdgcn_wmma_f32_16x16x32_f16(
            false, a0[j].v, false, bfrag[kt], (short)0, acc0, false, false);
        acc1 = __builtin_amdgcn_wmma_f32_16x16x32_f16(
            false, a1[j].v, false, bfrag[kt], (short)0, acc1, false, false);
      }
    }

    v8h hn0, hn1;
#pragma unroll
    for (int r = 0; r < 8; ++r) {
      const float v0 = fast_tanh((float)xp0[r] + acc0[r]);
      const float v1 = fast_tanh((float)xp1[r] + acc1[r]);
      pool0[r] += v0;  pool1[r] += v1;
      hn0[r] = (_Float16)v0;  hn1[r] = (_Float16)v1;
    }
#pragma unroll
    for (int r = 0; r < 8; ++r) {
      nxt[(8 * g + r) * LDH + n] = hn0[r];
      nxt[(16 + 8 * g + r) * LDH + n] = hn1[r];
    }
    __syncthreads();  // one barrier/step is sufficient with double buffering
  }

  float* pp = &pool[((size_t)d * H + n) * B];
#pragma unroll
  for (int r = 0; r < 8; ++r) {
    pp[8 * g + r] = pool0[r];
    pp[16 + 8 * g + r] = pool1[r];
  }
}

// ============ Kernel 4: mean -> tanh -> fc head (tiny) ============
__global__ void grd_head(const float* __restrict__ pool, const float* __restrict__ fcW,
                         const float* __restrict__ fcb, float* __restrict__ out) {
  const int t = blockIdx.x * blockDim.x + threadIdx.x;
  if (t >= B * G) return;
  const int b = t / G, gg = t % G;
  const float inv = 1.0f / (float)S;
  float sum = fcb[gg];
  for (int j = 0; j < H; ++j) {
    const float af = fast_tanh(pool[(size_t)(0 * H + j) * B + b] * inv);
    const float ab = fast_tanh(pool[(size_t)(1 * H + j) * B + b] * inv);
    sum += af * fcW[gg * 2 * H + j] + ab * fcW[gg * 2 * H + H + j];
  }
  out[(size_t)b * G + gg] = sum;
}

extern "C" void kernel_launch(void* const* d_in, const int* in_sizes, int n_in,
                              void* d_out, int out_size, void* d_ws, size_t ws_size,
                              hipStream_t stream) {
  const int*   tok  = (const int*)d_in[0];
  const float* emb  = (const float*)d_in[1];
  const float* WihF = (const float*)d_in[2];
  const float* WhhF = (const float*)d_in[3];
  const float* bihF = (const float*)d_in[4];
  const float* bhhF = (const float*)d_in[5];
  const float* WihB = (const float*)d_in[6];
  const float* WhhB = (const float*)d_in[7];
  const float* bihB = (const float*)d_in[8];
  const float* bhhB = (const float*)d_in[9];
  const float* fcW  = (const float*)d_in[10];
  const float* fcb  = (const float*)d_in[11];
  float* out = (float*)d_out;

  char* ws = (char*)d_ws;
  _Float16* wih16 = (_Float16*)(ws + OFF_WIH);
  _Float16* whh16 = (_Float16*)(ws + OFF_WHH);
  float*    bias  = (float*)(ws + OFF_BIAS);
  float*    pool  = (float*)(ws + OFF_POOL);
  _Float16* xproj = (_Float16*)(ws + OFF_XP);

  grd_pack_params<<<(2 * H * KP + 255) / 256, 256, 0, stream>>>(
      WihF, WhhF, bihF, bhhF, WihB, WhhB, bihB, bhhB, wih16, whh16, bias);

  grd_embed_proj<<<(S * B) / 16, 256, 0, stream>>>(tok, emb, wih16, bias, xproj);

  grd_rnn_scan<<<2, 512, 0, stream>>>(whh16, xproj, pool);

  grd_head<<<1, 192, 0, stream>>>(pool, fcW, fcb, out);
}
